// RecurrentNEFLayer_64287070486909
// MI455X (gfx1250) — matrix operand order repaired and checked by reading, not compiled, not run on hardware
//
#include <hip/hip_runtime.h>

// Problem constants (from reference)
#define BATCH  512
#define TLEN   256
#define DIN    64
#define NNEUR  2048
#define DSTATE 64
#define DAUG   128   // DIN + DSTATE
#define DOUT   10

typedef __attribute__((ext_vector_type(16))) __bf16 v16bf;
typedef __attribute__((ext_vector_type(8)))  __bf16 v8bf;
typedef __attribute__((ext_vector_type(8)))  float  v8f;

union AFrag { v16bf v; v8bf h[2]; };

// ---------------------------------------------------------------------------
// Prep: convert f32 weights to bf16 in WMMA-friendly layouts in workspace.
//   encB[n][k]   = (bf16) encoders[n][k]            (row-major, N x 128)
//   sdt [ns][n]  = (bf16) state_decoders[n][ns]     (transposed, 64 x 2048)
// Transposing SD makes the GEMM2 B-fragment a contiguous 32B load per lane.
// ---------------------------------------------------------------------------
__global__ void nef_prep_kernel(const float* __restrict__ enc,
                                const float* __restrict__ sd,
                                __bf16* __restrict__ encB,
                                __bf16* __restrict__ sdt) {
  int i = blockIdx.x * blockDim.x + threadIdx.x;
  if (i < NNEUR * DAUG) {
    encB[i] = (__bf16)enc[i];
  } else {
    int j = i - NNEUR * DAUG;
    if (j < DSTATE * NNEUR) {
      int ns = j >> 11;          // j / 2048
      int n  = j & (NNEUR - 1);  // j % 2048
      sdt[j] = (__bf16)sd[n * DSTATE + ns];
    }
  }
}

// ---------------------------------------------------------------------------
// Main recurrent kernel. One workgroup = 16 batch rows, 8 waves (wave32).
// Each wave owns 256 neurons (8 chunks of 32). Per timestep:
//   1) stage x_aug = [u_t, s] as bf16 in LDS; zero next-state buffer
//   2) GEMM1 per 16x16 tile: d = x_aug @ enc^T  (4x wmma_bf16, K=128)
//      epilogue a = |gain*d + bias|, bounce through per-wave LDS scratch
//      to repack C/D layout -> A layout
//   3) GEMM2: s_partial += a(16x32) @ SD(32x64)  (4x wmma_bf16 per chunk)
//   4) reduce s_partial across waves with ds_add_f32; barrier
// Final step also folds in the (2048 x 10) output decode per chunk.
// ---------------------------------------------------------------------------
__global__ __launch_bounds__(256)
void nef_recurrent_kernel(const float* __restrict__ seq,
                          const float* __restrict__ gain,
                          const float* __restrict__ bias,
                          const float* __restrict__ decoders,
                          const __bf16* __restrict__ encB,
                          const __bf16* __restrict__ sdt,
                          float* __restrict__ out) {
  __shared__ __align__(16) __bf16 xaug[16][DAUG];        // 4 KB
  __shared__ __align__(16) float  sbuf[2][16][DSTATE];   // 8 KB (double-buffered state)
  __shared__ __align__(16) __bf16 ascr[8][16][32];       // 8 KB (per-wave a-tile scratch)
  __shared__ __align__(16) float  gbuf[NNEUR];           // 8 KB
  __shared__ __align__(16) float  bbuf[NNEUR];           // 8 KB
  __shared__ __align__(16) float  outacc[16][DOUT];      // 640 B

  const int tid   = threadIdx.x;
  const int lane  = tid & 31;
  const int wave  = tid >> 5;
  const int lo16  = lane & 15;   // column (N) / row (M) index within tile
  const int hi    = lane >> 4;   // lane-half selects K sub-range
  const int bbase = blockIdx.x * 16;

  // Preload gain/bias into LDS; zero state buffer 0 and output accumulator.
  for (int i = tid; i < NNEUR; i += 256) { gbuf[i] = gain[i]; bbuf[i] = bias[i]; }
  for (int i = tid; i < 16 * DSTATE; i += 256) ((float*)sbuf[0])[i] = 0.f;
  for (int i = tid; i < 16 * DOUT;  i += 256) ((float*)outacc)[i]   = 0.f;
  __syncthreads();

  #pragma unroll 1
  for (int t = 0; t < TLEN; ++t) {
    const int r = t & 1;       // state buffer to read
    const int w = r ^ 1;       // state buffer to accumulate into

    // ---- Step 1: build x_aug (bf16) and zero next-state buffer ----
    {
      int idx = tid * 8;                 // 256 threads x 8 elems = 16*128
      int m = idx >> 7, c = idx & 127;
      if (c < DIN) {
        const float* sp = &seq[((size_t)(bbase + m) * TLEN + t) * DIN + c];
        #pragma unroll
        for (int j = 0; j < 8; ++j) xaug[m][c + j] = (__bf16)sp[j];
      } else {
        #pragma unroll
        for (int j = 0; j < 8; ++j) xaug[m][c + j] = (__bf16)sbuf[r][m][c - DIN + j];
      }
      for (int i = tid; i < 16 * DSTATE; i += 256) ((float*)sbuf[w])[i] = 0.f;
    }
    __syncthreads();

    // ---- A fragments for GEMM1 (16x32 bf16, contiguous K order), reused 16x ----
    // lanes 0-15: elems 0-15 = K 0-15 (M = lane)
    // lanes 16-31: elems 0-15 = K 16-31 (M = lane-16)
    AFrag A[4];
    #pragma unroll
    for (int ks = 0; ks < 4; ++ks) {
      A[ks].h[0] = *(const v8bf*)&xaug[lo16][ks * 32 + hi * 16];
      A[ks].h[1] = *(const v8bf*)&xaug[lo16][ks * 32 + hi * 16 + 8];
    }

    v8f sacc[4] = {};   // state partial: 16 x 64 across 4 N-tiles

    #pragma unroll 1
    for (int c8 = 0; c8 < 8; ++c8) {
      const int nbase = (wave * 8 + c8) * 32;

      #pragma unroll
      for (int h2 = 0; h2 < 2; ++h2) {
        const int n0 = nbase + h2 * 16;
        v8f d = {};
        #pragma unroll
        for (int ks = 0; ks < 4; ++ks) {
          // B fragment (32x16): lane = N (lo16); lane-half holds K 0-15 / 16-31
          v16bf Bf = *(const v16bf*)&encB[(size_t)(n0 + lo16) * DAUG + ks * 32 + hi * 16];
          d = __builtin_amdgcn_wmma_f32_16x16x32_bf16(
                false, A[ks].v, false, Bf, (short)0, d, false, false);
        }
        // Epilogue: a = |gain*d + bias|; D layout: lane holds col N=lo16, rows M=v+8*hi
        const float g  = gbuf[n0 + lo16];
        const float bi = bbuf[n0 + lo16];
        #pragma unroll
        for (int v = 0; v < 8; ++v) {
          float av = __builtin_fabsf(g * d[v] + bi);
          ascr[wave][v + 8 * hi][h2 * 16 + lo16] = (__bf16)av;  // repack via LDS
        }
      }

      // ---- GEMM2: sacc += a(16x32) @ SD(32x64). Same-wave LDS is in-order,
      // so the A2 loads below see the stores above without a barrier. ----
      AFrag A2;
      A2.h[0] = *(const v8bf*)&ascr[wave][lo16][hi * 16];
      A2.h[1] = *(const v8bf*)&ascr[wave][lo16][hi * 16 + 8];
      #pragma unroll
      for (int j2 = 0; j2 < 4; ++j2) {
        // B2[k][ns] = SD[nbase+k][j2*16+lo16] = sdt[(j2*16+lo16)*2048 + nbase + hi*16 + i]
        v16bf B2 = *(const v16bf*)&sdt[(size_t)(j2 * 16 + lo16) * NNEUR + nbase + hi * 16];
        sacc[j2] = __builtin_amdgcn_wmma_f32_16x16x32_bf16(
                     false, A2.v, false, B2, (short)0, sacc[j2], false, false);
      }

      // Final timestep: fold this chunk's contribution into out = a_final @ dec
      if (t == TLEN - 1) {
        #pragma unroll 1
        for (int p = lane; p < 16 * DOUT; p += 32) {
          int m = p / DOUT, o = p - m * DOUT;
          float acc = 0.f;
          #pragma unroll
          for (int k = 0; k < 32; ++k)
            acc += (float)ascr[wave][m][k] * decoders[(nbase + k) * DOUT + o];
          atomicAdd(&outacc[m][o], acc);
        }
      }
    }

    // ---- Step 4: reduce state partials across waves (ds_add_f32) ----
    #pragma unroll
    for (int j2 = 0; j2 < 4; ++j2)
      #pragma unroll
      for (int v = 0; v < 8; ++v)
        atomicAdd(&sbuf[w][v + 8 * hi][j2 * 16 + lo16], sacc[j2][v]);

    __syncthreads();
  }

  // Write final output (B x 10, f32)
  for (int p = tid; p < 16 * DOUT; p += 256) {
    int m = p / DOUT, o = p - m * DOUT;
    out[(size_t)(bbase + m) * DOUT + o] = outacc[m][o];
  }
}

// ---------------------------------------------------------------------------
// Launch: prep (weight conversion into d_ws) then the recurrent kernel.
// d_in order: seq, encoders, bias, gain, state_decoders, decoders
// ---------------------------------------------------------------------------
extern "C" void kernel_launch(void* const* d_in, const int* in_sizes, int n_in,
                              void* d_out, int out_size, void* d_ws, size_t ws_size,
                              hipStream_t stream) {
  const float* seq  = (const float*)d_in[0];
  const float* enc  = (const float*)d_in[1];
  const float* bias = (const float*)d_in[2];
  const float* gain = (const float*)d_in[3];
  const float* sd   = (const float*)d_in[4];
  const float* dec  = (const float*)d_in[5];
  float* out = (float*)d_out;

  __bf16* encB = (__bf16*)d_ws;                                   // 512 KB
  __bf16* sdt  = (__bf16*)((char*)d_ws + (size_t)NNEUR * DAUG * 2); // 256 KB

  const int total = NNEUR * DAUG + DSTATE * NNEUR;
  nef_prep_kernel<<<(total + 255) / 256, 256, 0, stream>>>(enc, sd, encB, sdt);
  nef_recurrent_kernel<<<BATCH / 16, 256, 0, stream>>>(seq, gain, bias, dec,
                                                       encB, sdt, out);
}
